// Linear_39891656245349
// MI455X (gfx1250) — 8-bit matrix operand order repaired and checked by reading, compiled, not run on hardware
//
#include <hip/hip_runtime.h>
#include <stdint.h>

#define MDIM 8192
#define NDIM 4096
#define KDIM 4096
#define BLK  128
#define KB   (KDIM / BLK)

typedef __attribute__((ext_vector_type(16))) int   v16i;
typedef __attribute__((ext_vector_type(8)))  float v8f;
typedef __attribute__((ext_vector_type(4)))  int   v4i;
typedef __attribute__((ext_vector_type(4)))  float v4f;

__device__ __forceinline__ void prefetch_wgp(const void* p) {
  asm volatile("global_prefetch_b8 %0, off" :: "v"(p));
}

__device__ __forceinline__ unsigned f32_to_e4m3(float x) {
  unsigned u = __float_as_uint(x);
  unsigned s = (u >> 31) << 7;
  float a = __uint_as_float(u & 0x7fffffffu);
  a = fminf(a, 448.0f);
  if (a < 0.015625f) {
    float t = fmaf(a, 512.0f, 12582912.0f);
    unsigned q = __float_as_uint(t) & 0xffu;
    return s | q;
  }
  unsigned b = __float_as_uint(a);
  unsigned lsb = (b >> 20) & 1u;
  b += 0x7ffffu + lsb;
  unsigned code = (((b >> 23) - 120u) << 3) | ((b >> 20) & 7u);
  if (code > 0x7eu) code = 0x7eu;
  return s | code;
}

__global__ __launch_bounds__(256) void act_quant_kernel(
    const float* __restrict__ x, uint8_t* __restrict__ axq,
    float* __restrict__ xs_t) {
  const int t   = threadIdx.x;
  const int r   = t >> 4;
  const int l   = t & 15;
  const int kb  = blockIdx.x;
  const int mt  = blockIdx.y;
  const int row = mt * 16 + r;

  const float* px = x + (size_t)row * KDIM + (size_t)kb * BLK + l * 8;
  v4f v0 = *(const v4f*)px;
  v4f v1 = *(const v4f*)(px + 4);

  float amax = 0.0f;
#pragma unroll
  for (int j = 0; j < 4; ++j) {
    amax = fmaxf(amax, fabsf(v0[j]));
    amax = fmaxf(amax, fabsf(v1[j]));
  }
#pragma unroll
  for (int m = 1; m < 16; m <<= 1) amax = fmaxf(amax, __shfl_xor(amax, m, 32));

  float am    = fmaxf(amax, 1e-12f);
  float scale = am * (1.0f / 448.0f);
  float inv   = 448.0f / am;

  unsigned lo = 0, hi = 0;
#pragma unroll
  for (int j = 0; j < 4; ++j) lo |= f32_to_e4m3(v0[j] * inv) << (8 * j);
#pragma unroll
  for (int j = 0; j < 4; ++j) hi |= f32_to_e4m3(v1[j] * inv) << (8 * j);

  const int k0 = l * 8;
  const int h8 = (k0 >= 64) ? 8 : 0;
  const int k6 = k0 & 63;
  const int p  = k6 >> 4;
  const int h  = (k6 >> 3) & 1;
  const int vv = h8 + 2 * p;
  const int L  = h * 16 + r;

  uint8_t* tile = axq + ((size_t)mt * KB + kb) * 2048;
  const unsigned pos = (unsigned)((vv >> 2) * 512 + L * 16 + (vv & 3) * 4);
  *(uint2*)(tile + pos) = make_uint2(lo, hi);

  if (l == 0) xs_t[(size_t)kb * MDIM + row] = scale;
}

__global__ __launch_bounds__(256) void w_quant_kernel(
    const float* __restrict__ w, uint8_t* __restrict__ wq) {
  const int t  = threadIdx.x;
  const int r  = t >> 4;
  const int l  = t & 15;
  const int kb = blockIdx.x;
  const int nt = blockIdx.y;

  const float* pw = w + (size_t)(nt * 16 + r) * KDIM + (size_t)kb * BLK + l * 8;
  v4f v0 = *(const v4f*)pw;
  v4f v1 = *(const v4f*)(pw + 4);

  unsigned lo = 0, hi = 0;
#pragma unroll
  for (int j = 0; j < 4; ++j) lo |= f32_to_e4m3(v0[j]) << (8 * j);
#pragma unroll
  for (int j = 0; j < 4; ++j) hi |= f32_to_e4m3(v1[j]) << (8 * j);

  const int k0 = l * 8;
  const int g  = k0 >> 5;
  const int h  = (k0 >> 3) & 1;
  const int vm = ((k0 >> 4) & 1) * 2;
  const int L  = h * 16 + r;

  uint8_t* tile = wq + ((size_t)nt * KB + kb) * 2048;
  *(uint2*)(tile + (size_t)(g * 512 + L * 16 + vm * 4)) = make_uint2(lo, hi);
}

union Frag { v16i v; v4i q[4]; };

__global__ __launch_bounds__(256) void fp8_gemm_kernel(
    const uint8_t* __restrict__ axq, const uint8_t* __restrict__ wq,
    const float* __restrict__ xs_t, const float* __restrict__ wscale,
    float* __restrict__ out) {
  const int lane  = threadIdx.x & 31;
  const int wave  = threadIdx.x >> 5;
  const int waveM = wave >> 1;
  const int waveN = wave & 1;
  const int mt0   = blockIdx.y * 8 + waveM * 2;
  const int nt0   = blockIdx.x * 8 + waveN * 4;
  const int h     = lane >> 4;
  const int l16   = lane & 15;

  v8f acc[2][4];
#pragma unroll
  for (int mi = 0; mi < 2; ++mi)
#pragma unroll
    for (int ni = 0; ni < 4; ++ni) acc[mi][ni] = (v8f){};

  for (int kb = 0; kb < KB; ++kb) {
    Frag aF[2], bF[4];
#pragma unroll
    for (int mi = 0; mi < 2; ++mi) {
      const uint8_t* tp = axq + ((size_t)(mt0 + mi) * KB + kb) * 2048 + lane * 16;
#pragma unroll
      for (int i = 0; i < 4; ++i) aF[mi].q[i] = *(const v4i*)(tp + i * 512);
    }
#pragma unroll
    for (int ni = 0; ni < 4; ++ni) {
      const uint8_t* tp = wq + ((size_t)(nt0 + ni) * KB + kb) * 2048 + lane * 16;
#pragma unroll
      for (int i = 0; i < 4; ++i) bF[ni].q[i] = *(const v4i*)(tp + i * 512);
    }

    if (kb + 1 < KB) {
#pragma unroll
      for (int mi = 0; mi < 2; ++mi)
        prefetch_wgp(axq + ((size_t)(mt0 + mi) * KB + kb + 1) * 2048 + lane * 16);
#pragma unroll
      for (int ni = 0; ni < 4; ++ni)
        prefetch_wgp(wq + ((size_t)(nt0 + ni) * KB + kb + 1) * 2048 + lane * 16);
    }

    const float sw = wscale[(size_t)blockIdx.x * KB + kb];

    float sm[2][8];
#pragma unroll
    for (int mi = 0; mi < 2; ++mi) {
      const float* sp = xs_t + (size_t)kb * MDIM + (mt0 + mi) * 16 + (h << 3);
      v4f s0 = *(const v4f*)sp;
      v4f s1 = *(const v4f*)(sp + 4);
#pragma unroll
      for (int r = 0; r < 4; ++r) { sm[mi][r] = s0[r] * sw; sm[mi][r + 4] = s1[r] * sw; }
    }

    v8f zero = {};
#pragma unroll
    for (int mi = 0; mi < 2; ++mi)
#pragma unroll
      for (int ni = 0; ni < 4; ++ni) {
        v8f p = __builtin_amdgcn_wmma_f32_16x16x128_fp8_fp8(
            aF[mi].v, bF[ni].v, (short)0, zero, false, false);
#pragma unroll
        for (int r = 0; r < 8; ++r)
          acc[mi][ni][r] = fmaf(p[r], sm[mi][r], acc[mi][ni][r]);
      }
  }

#pragma unroll
  for (int mi = 0; mi < 2; ++mi)
#pragma unroll
    for (int ni = 0; ni < 4; ++ni) {
      const int col = (nt0 + ni) * 16 + l16;
#pragma unroll
      for (int r = 0; r < 8; ++r) {
        const int row = (mt0 + mi) * 16 + r + (h << 3);
        out[(size_t)row * NDIM + col] = acc[mi][ni][r];
      }
    }
}

extern "C" void kernel_launch(void* const* d_in, const int* in_sizes, int n_in,
                              void* d_out, int out_size, void* d_ws, size_t ws_size,
                              hipStream_t stream) {
  const float* x  = (const float*)d_in[0];
  const float* w  = (const float*)d_in[1];
  const float* ws = (const float*)d_in[2];
  float* out = (float*)d_out;

  uint8_t* axq = (uint8_t*)d_ws;
  uint8_t* wqp = axq + (size_t)MDIM * KDIM;
  float*   xst = (float*)(wqp + (size_t)NDIM * KDIM);

  act_quant_kernel<<<dim3(KB, MDIM / 16), 256, 0, stream>>>(x, axq, xst);
  w_quant_kernel<<<dim3(KB, NDIM / 16), 256, 0, stream>>>(w, wqp);
  fp8_gemm_kernel<<<dim3(NDIM / 128, MDIM / 128), 256, 0, stream>>>(
      axq, wqp, xst, ws, out);
}
